// HindiCausalLMAttention_82652350644313
// MI455X (gfx1250) — matrix operand order repaired and checked by reading, not compiled, not run on hardware
//
#include <hip/hip_runtime.h>
#include <hip/hip_bf16.h>

typedef _Float16 v8h  __attribute__((ext_vector_type(8)));
typedef _Float16 v16h __attribute__((ext_vector_type(16)));
typedef _Float16 h4   __attribute__((ext_vector_type(4)));
typedef float    v8f  __attribute__((ext_vector_type(8)));
typedef unsigned int u32x4 __attribute__((ext_vector_type(4)));
typedef int          i32x4 __attribute__((ext_vector_type(4)));
typedef int          i32x8 __attribute__((ext_vector_type(8)));

#define WMMA_F16(A,B,C) \
  __builtin_amdgcn_wmma_f32_16x16x32_f16(false,(A),false,(B),(short)0,(C),false,false)

static __device__ __forceinline__ v16h cat8(v8h a, v8h b) {
  v16h r;
#pragma unroll
  for (int i = 0; i < 8; ++i) { r[i] = a[i]; r[i + 8] = b[i]; }
  return r;
}

// ---------------------------------------------------------------------------
// Tensor Data Mover: 2D tile load (128x128 f16, row stride in elements) into
// LDS.  D# groups built in scalar regs; tracked by TENSORcnt.
// ---------------------------------------------------------------------------
#if defined(__has_builtin)
#if __has_builtin(__builtin_amdgcn_tensor_load_to_lds)
#define HAVE_TDM 1
#endif
#endif

#ifdef HAVE_TDM
static __device__ __forceinline__ void tdm_load_2d_128x128_f16(
    unsigned lds_off, const void* gaddr, unsigned stride_elems)
{
  unsigned long long ga = (unsigned long long)gaddr;
  u32x4 g0;
  g0[0] = 1u;                                       // count=1 (valid D#)
  g0[1] = lds_off;                                  // lds_addr (bytes)
  g0[2] = (unsigned)(ga & 0xffffffffu);             // global_addr[31:0]
  g0[3] = (unsigned)((ga >> 32) & 0x1ffffffu)       // global_addr[56:32]
        | (2u << 30);                               // type = 2 ("image")
  i32x8 g1;
  g1[0] = (int)(1u << 16);                          // wg_mask=0, data_size=1 (2B)
  g1[1] = (int)(128u << 16);                        // tensor_dim0 = 128
  g1[2] = (int)(128u << 16);                        // tensor_dim1 = 128
  g1[3] = (int)(128u << 16);                        // tile_dim0 = 128
  g1[4] = (int)128;                                 // tile_dim1 = 128, tile_dim2 = 0
  g1[5] = (int)stride_elems;                        // tensor_dim0_stride[31:0]
  g1[6] = 0;                                        // stride hi, dim1_stride lo
  g1[7] = 0;
  i32x4 z4 = {0, 0, 0, 0};
#if __clang_major__ >= 23
  i32x8 z8 = {0, 0, 0, 0, 0, 0, 0, 0};
  __builtin_amdgcn_tensor_load_to_lds(g0, g1, z4, z4, z8, 0);
#else
  __builtin_amdgcn_tensor_load_to_lds(g0, g1, z4, z4, 0);
#endif
}
#endif

// ---------------------------------------------------------------------------
// Tiled GEMM: C[M,N] = A[M,K] * W[K,N].  A f32 or f16 (converted to f16 in LDS
// staging), W f32 (converted in staging), f32 WMMA accum.  Block tile
// 256x128x64, 256 threads = 8 wave32 waves (4 x 2), each wave 64x64
// (16 WMMAs per 8 fragment loads per k-step).
// TRANSV: store C transposed per-batch as vt[b][col][s] (for the V projection)
// using contiguous v8h stores (C-fragment rows are consecutive in M).
// ---------------------------------------------------------------------------
template <typename AT, typename OT, bool TRANSV = false>
__global__ __launch_bounds__(256)
void gemm_tn(const AT* __restrict__ A, const float* __restrict__ W,
             OT* __restrict__ C, int M, int N, int K)
{
  constexpr int BM = 256, BN = 128, BK = 64;
  __shared__ _Float16 As[BM][BK];   // [m][k]                    32 KB
  __shared__ _Float16 Bs[BN][BK];   // [n][k]  (W transposed)    16 KB
  const int t    = threadIdx.x;
  const int lane = t & 31;
  const int w    = t >> 5;
  const int wm   = w >> 1;      // 0..3  (64 rows each)
  const int wn   = w & 1;       // 0..1  (64 cols each)
  const int lm   = lane & 15;
  const int hi   = lane >> 4;
  const int m0   = blockIdx.y * BM;
  const int n0   = blockIdx.x * BN;

  v8f acc[4][4] = {};

  for (int k0 = 0; k0 < K; k0 += BK) {
    __syncthreads();
    // ---- stage A tile (convert to f16) ----
    if (sizeof(AT) == 4) {
#pragma unroll
      for (int l = 0; l < 16; ++l) {
        int e  = (l * 256 + t) * 4;            // 0..16383 over [BM][BK]
        int ko = e & (BK - 1);
        int mo = e >> 6;
        float4 f = *(const float4*)((const float*)A + (size_t)(m0 + mo) * K + k0 + ko);
        h4 h; h[0] = (_Float16)f.x; h[1] = (_Float16)f.y;
        h[2] = (_Float16)f.z; h[3] = (_Float16)f.w;
        *(h4*)&As[mo][ko] = h;
      }
    } else {
#pragma unroll
      for (int l = 0; l < 8; ++l) {
        int e  = (l * 256 + t) * 8;
        int ko = e & (BK - 1);
        int mo = e >> 6;
        *(v8h*)&As[mo][ko] =
            *(const v8h*)((const _Float16*)A + (size_t)(m0 + mo) * K + k0 + ko);
      }
    }
    // ---- stage W tile transposed (convert to f16) ----
#pragma unroll
    for (int l = 0; l < 8; ++l) {
      int e  = (l * 256 + t) * 4;              // 0..8191 over [BK][BN]
      int no = e & (BN - 1);
      int ko = e >> 7;
      float4 f = *(const float4*)(W + (size_t)(k0 + ko) * N + n0 + no);
      Bs[no + 0][ko] = (_Float16)f.x;
      Bs[no + 1][ko] = (_Float16)f.y;
      Bs[no + 2][ko] = (_Float16)f.z;
      Bs[no + 3][ko] = (_Float16)f.w;
    }
    __syncthreads();
    // ---- 2 WMMA k-steps of 32 ----
#pragma unroll
    for (int s = 0; s < 2; ++s) {
      const int ks = s * 32;
      v16h af[4], bf[4];
#pragma unroll
      for (int i = 0; i < 4; ++i) {           // A frag: K halves hi*8..+7, 16+hi*8..+7
        const _Float16* p = &As[wm * 64 + i * 16 + lm][ks + hi * 8];
        af[i] = cat8(*(const v8h*)p, *(const v8h*)(p + 16));
      }
#pragma unroll
      for (int jn = 0; jn < 4; ++jn) {        // B frag: K halves hi*8..+7, 16+hi*8..+7
        const _Float16* p = &Bs[wn * 64 + jn * 16 + lm][ks + hi * 8];
        bf[jn] = cat8(*(const v8h*)p, *(const v8h*)(p + 16));
      }
#pragma unroll
      for (int i = 0; i < 4; ++i)
#pragma unroll
        for (int jn = 0; jn < 4; ++jn)
          acc[i][jn] = WMMA_F16(af[i], bf[jn], acc[i][jn]);
    }
  }
  // ---- epilogue: C layout VGPR r <-> M = r + 8*hi, lane lm <-> N ----
  if (TRANSV) {
    // vt[(b*2048 + col) * 2048 + s], rows r are 8 consecutive s values
#pragma unroll
    for (int i = 0; i < 4; ++i) {
      int row0 = m0 + wm * 64 + i * 16 + hi * 8;
      int b    = row0 >> 11;
      int s0   = row0 & 2047;
#pragma unroll
      for (int jn = 0; jn < 4; ++jn) {
        int col = n0 + wn * 64 + jn * 16 + lm;
        v8h pk;
#pragma unroll
        for (int r = 0; r < 8; ++r) pk[r] = (_Float16)acc[i][jn][r];
        *(v8h*)((_Float16*)C + ((size_t)(b * 2048 + col)) * 2048 + s0) = pk;
      }
    }
  } else {
#pragma unroll
    for (int i = 0; i < 4; ++i)
#pragma unroll
      for (int jn = 0; jn < 4; ++jn)
#pragma unroll
        for (int r = 0; r < 8; ++r) {
          int row = m0 + wm * 64 + i * 16 + hi * 8 + r;
          int col = n0 + wn * 64 + jn * 16 + lm;
          C[(size_t)row * N + col] = (OT)acc[i][jn][r];
        }
  }
}

// ---------------------------------------------------------------------------
// RoPE in-place on f16 Q or K laid out [B*S, nh*hd].  position_ids == arange.
// ---------------------------------------------------------------------------
__global__ __launch_bounds__(256)
void rope_kernel(_Float16* __restrict__ x, const float* __restrict__ ct,
                 const float* __restrict__ st)
{
  const int S = 2048, H = 2048;
  int tid  = blockIdx.x * 256 + threadIdx.x;   // B*S*16*64 threads
  int d    = tid & 63;
  int hidx = (tid >> 6) & 15;
  int row  = tid >> 10;                        // b*S + s
  int s    = row & (S - 1);
  size_t base = (size_t)row * H + hidx * 128;
  float x0 = (float)x[base + d];
  float x1 = (float)x[base + d + 64];
  float c0 = ct[s * 128 + d],  c1 = ct[s * 128 + d + 64];
  float s0 = st[s * 128 + d],  s1 = st[s * 128 + d + 64];
  x[base + d]      = (_Float16)(x0 * c0 - x1 * s0);
  x[base + d + 64] = (_Float16)(x1 * c1 + x0 * s1);
}

// ---------------------------------------------------------------------------
// Flash attention: one block per (q-tile of 128, head, batch), 8 waves, each
// wave owns 16 query rows.  K and V^T tiles are DMA'd into LDS by the Tensor
// Data Mover (V^T comes pre-transposed from the V-projection epilogue), with
// TENSORcnt/S_WAIT_TENSORCNT synchronization.  Online softmax; causal mask
// folded analytically.  KP holds the K tile, then is reused for the P tile.
// ---------------------------------------------------------------------------
__global__ __launch_bounds__(256)
void flash_attn(const _Float16* __restrict__ q, const _Float16* __restrict__ k,
                const _Float16* __restrict__ vt, _Float16* __restrict__ o)
{
  const int S = 2048, H = 2048, HD = 128;
  __shared__ _Float16 KP[128 * 128];   // K tile [key][dim], later P [qrow][key]
  __shared__ _Float16 VT[128 * 128];   // V^T tile [dim][key]
  const int t    = threadIdx.x;
  const int lane = t & 31;
  const int w    = t >> 5;
  const int lm   = lane & 15;
  const int hi   = lane >> 4;
  const int qt = blockIdx.x, h = blockIdx.y, b = blockIdx.z;
  const size_t rb = (size_t)b * S;

  // Q fragments for this wave's 16 rows stay in registers for the whole loop.
  v16h aq[4];
  {
    const _Float16* qp = q + (rb + qt * 128 + w * 16 + lm) * H + h * HD;
#pragma unroll
    for (int c = 0; c < 4; ++c)
      aq[c] = cat8(*(const v8h*)(qp + c * 32 + hi * 8),
                   *(const v8h*)(qp + c * 32 + 16 + hi * 8));
  }

  float mrow[8], lrow[8];
  v8f oacc[8] = {};
#pragma unroll
  for (int r = 0; r < 8; ++r) { mrow[r] = -1e30f; lrow[r] = 0.f; }

  const float scale = 0.08838834764831845f;  // 1/sqrt(128)
#ifdef HAVE_TDM
  const unsigned ldsK = (unsigned)(unsigned long long)(void*)&KP[0];
  const unsigned ldsV = (unsigned)(unsigned long long)(void*)&VT[0];
#endif

  for (int j = 0; j <= qt; ++j) {
    const _Float16* kbase = k  + (rb + (size_t)j * 128) * H + h * HD;
    const _Float16* vbase = vt + ((size_t)b * 2048 + h * 128) * 2048 + j * 128;
    __syncthreads();   // previous iteration's KP/VT readers done
#ifdef HAVE_TDM
    if (w == 0) {      // one wave drives the DMA engine
      tdm_load_2d_128x128_f16(ldsK, kbase, (unsigned)H);
      tdm_load_2d_128x128_f16(ldsV, vbase, 2048u);
      __builtin_amdgcn_s_wait_tensorcnt(0);
    }
#else
#pragma unroll
    for (int l = 0; l < 8; ++l) {
      int e = l * 256 + t;               // 0..2047, 8 halves each
      int r = e >> 4;                    // 0..127
      int c = (e & 15) << 3;             // 0..120
      *(v8h*)&KP[r * 128 + c] = *(const v8h*)(kbase + (size_t)r * H + c);
      *(v8h*)&VT[r * 128 + c] = *(const v8h*)(vbase + (size_t)r * 2048 + c);
    }
#endif
    __syncthreads();
    // ---- scores S = Q * K^T ----
    v8f sacc[8];
#pragma unroll
    for (int nt = 0; nt < 8; ++nt) {
      v8f z = {}; sacc[nt] = z;
#pragma unroll
      for (int c = 0; c < 4; ++c) {
        const _Float16* p = &KP[(nt * 16 + lm) * 128 + c * 32 + hi * 8];
        v16h bf = cat8(*(const v8h*)p, *(const v8h*)(p + 16));
        sacc[nt] = WMMA_F16(aq[c], bf, sacc[nt]);
      }
    }
    // ---- scale + causal mask ----
    const int qi0 = qt * 128 + w * 16 + hi * 8;
#pragma unroll
    for (int nt = 0; nt < 8; ++nt) {
      const int kj = j * 128 + nt * 16 + lm;
#pragma unroll
      for (int r = 0; r < 8; ++r) {
        float sv = sacc[nt][r] * scale;
        sacc[nt][r] = (kj > qi0 + r) ? -1e9f : sv;
      }
    }
    __syncthreads();   // everyone finished reading K tile; KP now holds P
    // ---- online softmax ----
    float mx[8];
#pragma unroll
    for (int r = 0; r < 8; ++r) {
      float m = sacc[0][r];
#pragma unroll
      for (int nt = 1; nt < 8; ++nt) m = fmaxf(m, sacc[nt][r]);
#pragma unroll
      for (int off = 8; off >= 1; off >>= 1)
        m = fmaxf(m, __shfl_xor(m, off, 32));
      mx[r] = m;
    }
    float al[8];
#pragma unroll
    for (int r = 0; r < 8; ++r) {
      float mn = fmaxf(mrow[r], mx[r]);
      al[r]   = __expf(mrow[r] - mn);
      mrow[r] = mn;
    }
    float sum[8];
#pragma unroll
    for (int r = 0; r < 8; ++r) sum[r] = 0.f;
#pragma unroll
    for (int nt = 0; nt < 8; ++nt)
#pragma unroll
      for (int r = 0; r < 8; ++r) {
        float p = __expf(sacc[nt][r] - mrow[r]);
        sum[r] += p;
        KP[(w * 16 + hi * 8 + r) * 128 + nt * 16 + lm] = (_Float16)p;
      }
#pragma unroll
    for (int r = 0; r < 8; ++r) {
#pragma unroll
      for (int off = 8; off >= 1; off >>= 1)
        sum[r] += __shfl_xor(sum[r], off, 32);
      lrow[r] = lrow[r] * al[r] + sum[r];
    }
#pragma unroll
    for (int nt = 0; nt < 8; ++nt)
#pragma unroll
      for (int r = 0; r < 8; ++r) oacc[nt][r] *= al[r];
    // ---- O += P * V  (P read back from own rows: same-wave LDS in-order) ----
    v16h ap[4];
#pragma unroll
    for (int c = 0; c < 4; ++c) {
      const _Float16* p = &KP[(w * 16 + lm) * 128 + c * 32];
      ap[c] = cat8(*(const v8h*)(p + hi * 8), *(const v8h*)(p + 16 + hi * 8));
    }
#pragma unroll
    for (int nt = 0; nt < 8; ++nt)
#pragma unroll
      for (int c = 0; c < 4; ++c) {
        const _Float16* p = &VT[(nt * 16 + lm) * 128 + c * 32 + hi * 8];
        v16h bf = cat8(*(const v8h*)p, *(const v8h*)(p + 16));
        oacc[nt] = WMMA_F16(ap[c], bf, oacc[nt]);
      }
  }
  // ---- normalize and store ----
  _Float16* op = o + (rb + qt * 128 + w * 16 + hi * 8) * (size_t)H + h * HD;
#pragma unroll
  for (int r = 0; r < 8; ++r) {
    float inv = 1.f / lrow[r];
#pragma unroll
    for (int nt = 0; nt < 8; ++nt)
      op[(size_t)r * H + nt * 16 + lm] = (_Float16)(oacc[nt][r] * inv);
  }
}

// ---------------------------------------------------------------------------
extern "C" void kernel_launch(void* const* d_in, const int* in_sizes, int n_in,
                              void* d_out, int out_size, void* d_ws, size_t ws_size,
                              hipStream_t stream)
{
  (void)in_sizes; (void)n_in; (void)out_size; (void)ws_size;
  const float* hidden = (const float*)d_in[0];
  const float* cosT   = (const float*)d_in[1];
  const float* sinT   = (const float*)d_in[2];
  // d_in[3] position_ids == arange(S); d_in[4] attention_mask == causal:
  // both folded analytically into rope_kernel / flash_attn.
  const float* Wq = (const float*)d_in[5];
  const float* Wk = (const float*)d_in[6];
  const float* Wv = (const float*)d_in[7];
  const float* Wo = (const float*)d_in[8];
  float* out = (float*)d_out;

  const int B = 2, S = 2048, H = 2048;
  const size_t NE = (size_t)B * S * H;
  _Float16* qf = (_Float16*)d_ws;      // Q  (f16, [B*S, H])
  _Float16* kf = qf + NE;              // K  (f16, [B*S, H])
  _Float16* vtb = kf + NE;             // V^T (f16, [B, H, S])
  _Float16* of = vtb + NE;             // attention output (f16, [B*S, H])

  dim3 blk(256);
  dim3 gg(H / 128, (B * S) / 256);     // (16, 16)

  gemm_tn<float, _Float16, false><<<gg, blk, 0, stream>>>(hidden, Wq, qf, B * S, H, H);
  gemm_tn<float, _Float16, false><<<gg, blk, 0, stream>>>(hidden, Wk, kf, B * S, H, H);
  gemm_tn<float, _Float16, true ><<<gg, blk, 0, stream>>>(hidden, Wv, vtb, B * S, H, H);

  const int ropeBlocks = (B * S * 16 * 64) / 256;
  rope_kernel<<<ropeBlocks, blk, 0, stream>>>(qf, cosT, sinT);
  rope_kernel<<<ropeBlocks, blk, 0, stream>>>(kf, cosT, sinT);

  flash_attn<<<dim3(S / 128, 16, B), blk, 0, stream>>>(qf, kf, vtb, of);

  gemm_tn<_Float16, float, false><<<gg, blk, 0, stream>>>(of, Wo, out, B * S, H, H);
}
